// MashKLAutoEncoder_88811333746786
// MI455X (gfx1250) — matrix operand order repaired and checked by reading, not compiled, not run on hardware
//
#include <hip/hip_runtime.h>
#include <math.h>
#include <stdint.h>

// ---------------- model constants ----------------
#define BB   4
#define LL   400
#define TT   (BB * LL)          // 1600 tokens
#define DM   256                // d_model
#define DI   512                // d_inner
#define DST  16                 // d_state
#define NLAYER 24

#define USE_TDM 1

typedef _Float16 half8 __attribute__((ext_vector_type(8)));
typedef _Float16 v16h  __attribute__((ext_vector_type(16)));
typedef float    v8f   __attribute__((ext_vector_type(8)));
typedef unsigned int u32x4 __attribute__((ext_vector_type(4)));
typedef int          i32x4 __attribute__((ext_vector_type(4)));
typedef int          i32x8 __attribute__((ext_vector_type(8)));

__device__ __forceinline__ float sigm_(float x) { return 1.0f / (1.0f + __expf(-x)); }

// =========================================================================
// WMMA GEMM:  C = act(A @ op(B) + bias) (+ res)
//   A: [M,K] (row stride lda), f32
//   B: bkn==0 -> B is [N,K] (weight layout, row stride ldb) -> A @ B^T
//      bkn==1 -> B is [K,N] (row stride ldb)                -> A @ B
//   act: 0 none, 1 silu, 2 softplus, 3 (*= alpha)
//   batched via blockIdx.z with element strides sA/sB/sC (res shares sC).
// 128 threads = 4 wave32s; block tile 32x32; K step 32; f16 in, f32 accum.
// B tile staged as f32 into padded LDS; full bkn==0 tiles staged by the
// Tensor Data Mover (tensor_load_to_lds with pad_enable -> 36-DWORD rows).
// =========================================================================
__global__ __launch_bounds__(128) void wmma_gemm_kernel(
    const float* __restrict__ A, const float* __restrict__ B,
    const float* __restrict__ bias, const float* __restrict__ res,
    float* __restrict__ C,
    int M, int N, int K, int lda, int ldb, int ldc,
    int bkn, int act, float alpha,
    long long sA, long long sB, long long sC)
{
  A += (long long)blockIdx.z * sA;
  B += (long long)blockIdx.z * sB;
  C += (long long)blockIdx.z * sC;
  if (res) res += (long long)blockIdx.z * sC;

  __shared__ __align__(16) _Float16 As[32][40];  // f16, padded rows (80B)
  __shared__ __align__(16) float    Bs[32][36];  // f32 Bs[n][k], padded rows (144B)

  const int tid  = threadIdx.x;
  const int wave = tid >> 5;
  const int lane = tid & 31;
  const int h    = lane >> 4;          // half-wave select
  const int q    = lane & 15;
  const int wr   = (wave >> 1) << 4;   // wave subtile row (0/16)
  const int wc   = (wave & 1) << 4;    // wave subtile col (0/16)
  const int rb   = blockIdx.x * 32;
  const int cb   = blockIdx.y * 32;

  v8f acc = {0.f, 0.f, 0.f, 0.f, 0.f, 0.f, 0.f, 0.f};

  for (int k0 = 0; k0 < K; k0 += 32) {
    const bool b_full = (cb + 32 <= N) && (k0 + 32 <= K);
    const bool a_full = (rb + 32 <= M) && (k0 + 32 <= K);

    // ---------------- stage B tile: Bs[n][k] (f32) ----------------
#if USE_TDM
    const bool use_tdm = (bkn == 0) && b_full;
    if (use_tdm) {
      if (wave == 0) {
        // Build Tensor DMA Descriptor (D#) per CDNA5 ISA 8.3-8.6:
        // 2D tile 32(k) x 32(n), data_size=4B, dim0 stride = ldb,
        // LDS pad: every 32 DWORDs insert 4 DWORDs -> row stride 36 DWORDs.
        unsigned long long ga =
            (unsigned long long)(uintptr_t)(B + (long long)cb * ldb + k0);
        unsigned lds = (unsigned)(uintptr_t)(void*)&Bs[0][0];
        u32x4 g0; i32x8 g1;
        i32x4 g2 = {0, 0, 0, 0};
        i32x4 g3 = {0, 0, 0, 0};
        g0[0] = 1u;                                   // count=1, user mode
        g0[1] = lds;                                  // lds_addr (bytes)
        g0[2] = (unsigned)(ga & 0xFFFFFFFFu);         // global_addr[31:0]
        g0[3] = (unsigned)((ga >> 32) & 0x01FFFFFFu)  // global_addr[56:32]
              | (2u << 30);                           // type=2 ("image")
        const unsigned td0 = 32u, td1 = 32u, tile0 = 32u, tile1 = 32u;
        const unsigned long long s0 = (unsigned long long)(unsigned)ldb;
        g1[0] = (int)((2u << 16)      // data_size = 4 bytes
              | (1u << 20)            // pad_enable
              | (4u << 22)            // pad_interval: 32 DWORDs
              | (3u << 25));          // pad_amount: 4 DWORDs
        g1[1] = (int)((td0 & 0xFFFFu) << 16);                    // tensor_dim0 lo
        g1[2] = (int)((td0 >> 16) | ((td1 & 0xFFFFu) << 16));    // td0 hi | td1 lo
        g1[3] = (int)((td1 >> 16) | (tile0 << 16));              // td1 hi | tile_dim0
        g1[4] = (int)tile1;                                      // tile_dim1 (tile_dim2=0)
        g1[5] = (int)(unsigned)(s0 & 0xFFFFFFFFull);             // dim0_stride lo
        g1[6] = (int)(unsigned)((s0 >> 32) & 0xFFFFull);         // dim0_stride hi
        g1[7] = 0;
#if defined(__clang_major__) && (__clang_major__ >= 23)
        i32x8 g4 = {0, 0, 0, 0, 0, 0, 0, 0};
        __builtin_amdgcn_tensor_load_to_lds(g0, g1, g2, g3, g4, 0);
#else
        __builtin_amdgcn_tensor_load_to_lds(g0, g1, g2, g3, 0);
#endif
      }
    } else
#endif
    {
      if (bkn == 0) {
        if (b_full) {
          int n = tid >> 2, ks = (tid & 3) << 3;
          const float4* src = (const float4*)&B[(long long)(cb + n) * ldb + k0 + ks];
          float4 f0 = src[0], f1 = src[1];
          *(float4*)&Bs[n][ks]     = f0;
          *(float4*)&Bs[n][ks + 4] = f1;
        } else {
#pragma unroll
          for (int j = 0; j < 8; ++j) {
            int e = tid + 128 * j;
            int n = e >> 5, kk = e & 31;
            float v = 0.f;
            if (cb + n < N && k0 + kk < K) v = B[(long long)(cb + n) * ldb + k0 + kk];
            Bs[n][kk] = v;
          }
        }
      } else {
#pragma unroll
        for (int j = 0; j < 8; ++j) {
          int e = tid + 128 * j;
          int kk = e >> 5, n = e & 31;
          float v = 0.f;
          if (cb + n < N && k0 + kk < K) v = B[(long long)(k0 + kk) * ldb + cb + n];
          Bs[n][kk] = v;
        }
      }
    }

    // ---------------- stage A tile: As[m][k] (f16) ----------------
    if (a_full) {
      int r = tid >> 2, ks = (tid & 3) << 3;
      const float4* src = (const float4*)&A[(long long)(rb + r) * lda + k0 + ks];
      float4 f0 = src[0], f1 = src[1];
      half8 h8;
      h8[0] = (_Float16)f0.x; h8[1] = (_Float16)f0.y;
      h8[2] = (_Float16)f0.z; h8[3] = (_Float16)f0.w;
      h8[4] = (_Float16)f1.x; h8[5] = (_Float16)f1.y;
      h8[6] = (_Float16)f1.z; h8[7] = (_Float16)f1.w;
      *(half8*)&As[r][ks] = h8;
    } else {
#pragma unroll
      for (int j = 0; j < 8; ++j) {
        int e = tid + 128 * j;
        int r = e >> 5, kk = e & 31;
        float v = 0.f;
        if (rb + r < M && k0 + kk < K) v = A[(long long)(rb + r) * lda + k0 + kk];
        As[r][kk] = (_Float16)v;
      }
    }

    // prefetch next A tile (speculative; gfx1250 global_prefetch_b8)
    if (k0 + 32 < K && rb + lane < M)
      __builtin_prefetch(&A[(long long)(rb + lane) * lda + k0 + 32], 0, 1);

#if USE_TDM
    if (use_tdm && wave == 0) __builtin_amdgcn_s_wait_tensorcnt(0);
#endif
    __syncthreads();

    // ---- gather fragments per ISA 7.12.2 (wave32) ----
    // A (16x32 f16): lane m=q, element e -> k = (e/8)*16 + h*8 + e%8
    half8 a_lo = *(const half8*)&As[wr + q][h * 8];
    half8 a_hi = *(const half8*)&As[wr + q][16 + h * 8];
    v16h af = __builtin_shufflevector(a_lo, a_hi,
        0,1,2,3,4,5,6,7,8,9,10,11,12,13,14,15);
    // B (32x16 f16): lane n=q, element e -> k = (e/8)*16 + h*8 + e%8  (cvt from f32 LDS)
    const float* bp = &Bs[wc + q][h * 8];
    v16h bf;
#pragma unroll
    for (int e = 0; e < 16; ++e) bf[e] = (_Float16)bp[(e >> 3) * 16 + (e & 7)];

    acc = __builtin_amdgcn_wmma_f32_16x16x32_f16(
        false, af, false, bf, (short)0, acc, false, false);
    __syncthreads();
  }

  // ---- epilogue: C/D layout lane n=q, VGPR v -> m = v + 8*h ----
#pragma unroll
  for (int v = 0; v < 8; ++v) {
    int row = rb + wr + v + 8 * h;
    int col = cb + wc + q;
    if (row < M && col < N) {
      float val = acc[v];
      if (bias) val += bias[col];
      if (act == 1)      val = val * sigm_(val);                                   // silu
      else if (act == 2) val = (val > 20.f) ? val : log1pf(__expf(val));           // softplus
      else if (act == 3) val *= alpha;                                             // scale
      if (res) val += res[(long long)row * ldc + col];
      C[(long long)row * ldc + col] = val;
    }
  }
}

// ---------------- point embedding ----------------
// which: 0 rot (c=3,F=8,off=0), 1 pos (c=3,F=8,off=3),
//        2 mask (c=7,F=3,off=6), 3 sh (c=9,F=2,off=13)
__global__ __launch_bounds__(128) void embed_kernel(
    const float* __restrict__ mash,
    const float* __restrict__ W0, const float* __restrict__ b0,
    const float* __restrict__ W1, const float* __restrict__ b1,
    const float* __restrict__ W2, const float* __restrict__ b2,
    const float* __restrict__ W3, const float* __restrict__ b3,
    float* __restrict__ pose_emb, float* __restrict__ shape_emb)
{
  const int t = blockIdx.x;
  const int which = blockIdx.y;
  const int cArr[4]   = {3, 3, 7, 9};
  const int FArr[4]   = {8, 8, 3, 2};
  const int offArr[4] = {0, 3, 6, 13};
  const float* Wt[4]  = {W0, W1, W2, W3};
  const float* bt[4]  = {b0, b1, b2, b3};

  const int c = cArr[which], F = FArr[which], off = offArr[which];
  const int cF = c * F, inf = 2 * cF + c;
  const float* x = mash + (long long)t * 22 + off;

  __shared__ float feat[64];
  int j = threadIdx.x;
  if (j < cF) {
    float p = x[j / F] * 3.14159265358979323846f * (float)(1 << (j % F));
    feat[j]      = sinf(p);
    feat[cF + j] = cosf(p);
  } else if (j < cF + c) {
    feat[2 * cF + (j - cF)] = x[j - cF];
  }
  __syncthreads();

  const float* W = Wt[which];
  float s = bt[which][threadIdx.x];
  for (int k = 0; k < inf; ++k) s += feat[k] * W[threadIdx.x * inf + k];
  float* outp = (which < 2) ? (pose_emb + (long long)t * DM + which * 128)
                            : (shape_emb + (long long)t * DM + (which - 2) * 128);
  outp[threadIdx.x] = s;
}

// ---------------- rmsnorm with pose residual (one wave / token) ----------
// r = h + p ; hn = rmsnorm(r) * w
__global__ __launch_bounds__(32) void rmsnorm_resid_kernel(
    const float* __restrict__ hh, const float* __restrict__ pp,
    const float* __restrict__ w, float* __restrict__ rr, float* __restrict__ hn)
{
  const long long t = blockIdx.x;
  const int lane = threadIdx.x;
  float v[8]; float ss = 0.f;
#pragma unroll
  for (int j = 0; j < 8; ++j) {
    int i = lane + 32 * j;
    float x = hh[t * DM + i] + pp[t * DM + i];
    v[j] = x; rr[t * DM + i] = x; ss += x * x;
  }
#pragma unroll
  for (int off = 16; off > 0; off >>= 1) ss += __shfl_xor(ss, off, 32);
  float sc = rsqrtf(ss * (1.0f / DM) + 1e-5f);
#pragma unroll
  for (int j = 0; j < 8; ++j) {
    int i = lane + 32 * j;
    hn[t * DM + i] = v[j] * sc * w[i];
  }
}

// ---------------- layernorm (one wave / token) ----------------
__global__ __launch_bounds__(32) void layernorm_kernel(
    const float* __restrict__ x, const float* __restrict__ w,
    const float* __restrict__ b, float* __restrict__ out)
{
  const long long t = blockIdx.x;
  const int lane = threadIdx.x;
  float v[8]; float s = 0.f, sq = 0.f;
#pragma unroll
  for (int j = 0; j < 8; ++j) {
    int i = lane + 32 * j;
    float xv = x[t * DM + i];
    v[j] = xv; s += xv; sq += xv * xv;
  }
#pragma unroll
  for (int off = 16; off > 0; off >>= 1) {
    s  += __shfl_xor(s, off, 32);
    sq += __shfl_xor(sq, off, 32);
  }
  float m = s * (1.0f / DM);
  float var = sq * (1.0f / DM) - m * m;
  float sc = rsqrtf(var + 1e-5f);
#pragma unroll
  for (int j = 0; j < 8; ++j) {
    int i = lane + 32 * j;
    out[t * DM + i] = (v[j] - m) * sc * w[i] + b[i];
  }
}

// ---------------- causal depthwise conv (K=4) + silu ----------------
__global__ void conv_silu_kernel(const float* __restrict__ xz,
    const float* __restrict__ cw, const float* __restrict__ cb,
    float* __restrict__ xc)
{
  int idx = blockIdx.x * blockDim.x + threadIdx.x;
  if (idx >= TT * DI) return;
  int t = idx / DI, d = idx % DI;
  int l = t % LL;
  float s = cb[d];
#pragma unroll
  for (int k = 0; k < 4; ++k) {
    int lk = l - 3 + k;
    if (lk >= 0) s += xz[(long long)(t - 3 + k) * (2 * DI) + d] * cw[d * 4 + k];
  }
  xc[idx] = s * sigm_(s);
}

// ---------------- selective scan, fused gating epilogue ----------------
// channel = (b,d); 16-state in registers; y = (scan + xc*D) * silu(z)
__global__ __launch_bounds__(256) void scan_kernel(
    const float* __restrict__ dtb, const float* __restrict__ dbc,
    const float* __restrict__ xc, const float* __restrict__ xz,
    const float* __restrict__ A_log, const float* __restrict__ Dp,
    float* __restrict__ y)
{
  int idx = blockIdx.x * blockDim.x + threadIdx.x;   // 0..2047
  int b = idx / DI, d = idx % DI;
  float Ad[DST], h[DST];
#pragma unroll
  for (int s = 0; s < DST; ++s) { Ad[s] = -__expf(A_log[d * DST + s]); h[s] = 0.f; }
  for (int l = 0; l < LL; ++l) {
    long long t = (long long)b * LL + l;
    float dtv = dtb[t * DI + d];
    float xv  = xc[t * DI + d];
    const float* Bc = dbc + t * 48 + 16;
    const float* Cc = dbc + t * 48 + 32;
    float acc = 0.f;
#pragma unroll
    for (int s = 0; s < DST; ++s) {
      float hv = h[s] * __expf(dtv * Ad[s]) + dtv * Bc[s] * xv;
      h[s] = hv;
      acc += hv * Cc[s];
    }
    float zv = xz[t * (2 * DI) + DI + d];
    y[t * DI + d] = (acc + xv * Dp[d]) * (zv * sigm_(zv));
  }
}

// ---------------- row softmax over 400 ----------------
__global__ __launch_bounds__(128) void softmax_kernel(float* __restrict__ sim)
{
  const long long row = blockIdx.x;
  float* x = sim + row * LL;
  __shared__ float red[128];
  int tid = threadIdx.x;
  float m = -1e30f;
  for (int i = tid; i < LL; i += 128) m = fmaxf(m, x[i]);
  red[tid] = m; __syncthreads();
  for (int o = 64; o > 0; o >>= 1) { if (tid < o) red[tid] = fmaxf(red[tid], red[tid + o]); __syncthreads(); }
  m = red[0]; __syncthreads();
  float s = 0.f;
  for (int i = tid; i < LL; i += 128) { float e = __expf(x[i] - m); x[i] = e; s += e; }
  red[tid] = s; __syncthreads();
  for (int o = 64; o > 0; o >>= 1) { if (tid < o) red[tid] += red[tid + o]; __syncthreads(); }
  float inv = 1.0f / red[0];
  for (int i = tid; i < LL; i += 128) x[i] *= inv;
}

// ---------------- GEGLU gate (tanh gelu) ----------------
__global__ void geglu_kernel(const float* __restrict__ ff1, float* __restrict__ gg)
{
  int idx = blockIdx.x * blockDim.x + threadIdx.x;
  if (idx >= TT * 1024) return;
  int t = idx >> 10, i = idx & 1023;
  float a = ff1[(long long)t * 2048 + i];
  float g = ff1[(long long)t * 2048 + 1024 + i];
  float ge = 0.5f * g * (1.f + tanhf(0.7978845608028654f * (g + 0.044715f * g * g * g)));
  gg[idx] = a * ge;
}

// ---------------- latent: mean/logvar/z + KL ----------------
__global__ void latent_kernel(const float* __restrict__ x,
    const float* __restrict__ mw, const float* __restrict__ mb,
    const float* __restrict__ lw, const float* __restrict__ lb,
    const float* __restrict__ noise, float* __restrict__ z, float* __restrict__ kl)
{
  int t = blockIdx.x * blockDim.x + threadIdx.x;
  if (t >= TT) return;
  const float* xr = x + (long long)t * DM;
  float m = mb[0], lv = lb[0];
  for (int k = 0; k < DM; ++k) { m += xr[k] * mw[k]; lv += xr[k] * lw[k]; }
  lv = fminf(fmaxf(lv, -30.f), 20.f);
  z[t] = m + __expf(0.5f * lv) * noise[t];
  float c = 0.5f * (m * m + __expf(lv) - 1.f - lv);
  atomicAdd(&kl[t / LL], c);
}

__global__ void zero_kernel(float* __restrict__ p, int n)
{ int i = blockIdx.x * blockDim.x + threadIdx.x; if (i < n) p[i] = 0.f; }

__global__ void proj_kernel(const float* __restrict__ z,
    const float* __restrict__ pw, const float* __restrict__ pb, float* __restrict__ h)
{
  int idx = blockIdx.x * blockDim.x + threadIdx.x;
  if (idx >= TT * DM) return;
  int t = idx >> 8, d = idx & 255;
  h[idx] = z[t] * pw[d] + pb[d];
}

__global__ void assemble_kernel(const float* __restrict__ mash,
    const float* __restrict__ sout, float* __restrict__ out)
{
  int idx = blockIdx.x * blockDim.x + threadIdx.x;
  if (idx >= TT * 22) return;
  int t = idx / 22, j = idx % 22;
  out[idx] = (j < 6) ? mash[(long long)t * 22 + j] : sout[(long long)t * 16 + (j - 6)];
}

// =========================================================================
// host side
// =========================================================================
static inline void gemm(hipStream_t s, const float* A, const float* B,
                        const float* bias, const float* res, float* C,
                        int M, int N, int K, int lda, int ldb, int ldc,
                        int bkn, int act, float alpha,
                        int batch, long long sA, long long sB, long long sC)
{
  dim3 g((M + 31) / 32, (N + 31) / 32, batch);
  wmma_gemm_kernel<<<g, 128, 0, s>>>(A, B, bias, res, C, M, N, K, lda, ldb, ldc,
                                     bkn, act, alpha, sA, sB, sC);
}

struct StackW {
  const float *norm_w_l, *in_w, *conv_w, *conv_b, *xproj_w,
              *dt_w, *dt_b, *A_log, *Dp, *out_w, *final_norm;
};

static void run_stack(hipStream_t st, const float* pose_emb, float* hbuf, float* rbuf,
                      float* hnbuf, float* xz, float* xc, float* dbc, float* dtb,
                      float* yb, const StackW& w)
{
  for (int l = 0; l < NLAYER; ++l) {
    rmsnorm_resid_kernel<<<TT, 32, 0, st>>>(hbuf, pose_emb, w.norm_w_l + (long long)l * DM, rbuf, hnbuf);
    // in_proj: [T,256] x [1024,256]^T -> xz [T,1024]
    gemm(st, hnbuf, w.in_w + (long long)l * 2 * DI * DM, nullptr, nullptr, xz,
         TT, 2 * DI, DM, DM, DM, 2 * DI, 0, 0, 0.f, 1, 0, 0, 0);
    conv_silu_kernel<<<(TT * DI + 255) / 256, 256, 0, st>>>(xz, w.conv_w + (long long)l * DI * 4,
                                                            w.conv_b + (long long)l * DI, xc);
    // xproj: [T,512] x [48,512]^T -> dbc [T,48]
    gemm(st, xc, w.xproj_w + (long long)l * 48 * DI, nullptr, nullptr, dbc,
         TT, 48, DI, DI, DI, 48, 0, 0, 0.f, 1, 0, 0, 0);
    // dt: softplus(dbc[:,:16] x [512,16]^T + dt_b) -> dtb [T,512]
    gemm(st, dbc, w.dt_w + (long long)l * DI * 16, w.dt_b + (long long)l * DI, nullptr, dtb,
         TT, DI, 16, 48, 16, DI, 0, 2, 0.f, 1, 0, 0, 0);
    scan_kernel<<<(BB * DI) / 256, 256, 0, st>>>(dtb, dbc, xc, xz,
        w.A_log + (long long)l * DI * DST, w.Dp + (long long)l * DI, yb);
    // out_proj: [T,512] x [256,512]^T -> hbuf [T,256]
    gemm(st, yb, w.out_w + (long long)l * DM * DI, nullptr, nullptr, hbuf,
         TT, DM, DI, DI, DI, DM, 0, 0, 0.f, 1, 0, 0, 0);
  }
  // final: hn = rmsnorm(h + r) * final_norm
  rmsnorm_resid_kernel<<<TT, 32, 0, st>>>(hbuf, rbuf, w.final_norm, rbuf, hnbuf);
}

static void run_attn(hipStream_t st, const float* hq, const float* ctx,
                     float* xn, float* cn, float* qb, float* kvb, float* simb,
                     float* pvb, float* outp, const float* res, const float* const* ca)
{
  // ca: ln_q_w, ln_q_b, ln_c_w, ln_c_b, q_w, kv_w, out_w, out_b
  layernorm_kernel<<<TT, 32, 0, st>>>(hq, ca[0], ca[1], xn);
  layernorm_kernel<<<TT, 32, 0, st>>>(ctx, ca[2], ca[3], cn);
  gemm(st, xn, ca[4], nullptr, nullptr, qb, TT, DM, DM, DM, DM, DM, 0, 0, 0.f, 1, 0, 0, 0);
  gemm(st, cn, ca[5], nullptr, nullptr, kvb, TT, 2 * DM, DM, DM, DM, 2 * DM, 0, 0, 0.f, 1, 0, 0, 0);
  // sim[b] = q[b] @ k[b]^T / 16
  gemm(st, qb, kvb, nullptr, nullptr, simb, LL, LL, DM, DM, 2 * DM, LL, 0, 3, 0.0625f,
       BB, (long long)LL * DM, (long long)LL * 2 * DM, (long long)LL * LL);
  softmax_kernel<<<BB * LL, 128, 0, st>>>(simb);
  // out[b] = P[b] @ v[b]
  gemm(st, simb, kvb + DM, nullptr, nullptr, pvb, LL, DM, LL, LL, 2 * DM, DM, 1, 0, 0.f,
       BB, (long long)LL * LL, (long long)LL * 2 * DM, (long long)LL * DM);
  gemm(st, pvb, ca[6], ca[7], res, outp, TT, DM, DM, DM, DM, DM, 0, 0, 0.f, 1, 0, 0, 0);
}

static void run_ff(hipStream_t st, const float* x, float* xn, float* ff1, float* gg,
                   float* outp, const float* const* ff)
{
  // ff: ln_w, ln_b, w1, b1, w2, b2
  layernorm_kernel<<<TT, 32, 0, st>>>(x, ff[0], ff[1], xn);
  gemm(st, xn, ff[2], ff[3], nullptr, ff1, TT, 2048, DM, DM, DM, 2048, 0, 0, 0.f, 1, 0, 0, 0);
  geglu_kernel<<<(TT * 1024 + 255) / 256, 256, 0, st>>>(ff1, gg);
  gemm(st, gg, ff[4], ff[5], x, outp, TT, DM, 1024, 1024, 1024, DM, 0, 0, 0.f, 1, 0, 0, 0);
}

extern "C" void kernel_launch(void* const* d_in, const int* in_sizes, int n_in,
                              void* d_out, int out_size, void* d_ws, size_t ws_size,
                              hipStream_t stream)
{
  (void)in_sizes; (void)out_size; (void)ws_size;
  const float* P[80];
  for (int i = 0; i < n_in && i < 80; ++i) P[i] = (const float*)d_in[i];
  // d_in order (recursive dict insertion order of setup_inputs):
  //  0 mash
  //  1-3 rot basis/W/b, 4-6 pos, 7-9 mask, 10-12 sh
  //  13..22 enc_layers: norm_w,in_w,conv_w,conv_b,xproj_w,dt_w,dt_b,A_log,D,out_w
  //  23 enc_norm_w, 24..31 enc_ca, 32..37 enc_ff
  //  38 mean_w 39 mean_b 40 logvar_w 41 logvar_b 42 proj_w 43 proj_b
  //  44..53 dec_layers, 54 dec_norm_w, 55..62 dec_ca, 63..68 dec_ff
  //  69 out_w, 70 out_b, 71 noise

  float* w = (float*)d_ws;
  auto alloc = [&](size_t n) { float* p = w; w += n; return p; };
  float* pose_emb = alloc((size_t)TT * DM);
  float* hbuf  = alloc((size_t)TT * DM);
  float* rbuf  = alloc((size_t)TT * DM);
  float* hnbuf = alloc((size_t)TT * DM);
  float* xz    = alloc((size_t)TT * 2 * DI);
  float* xc    = alloc((size_t)TT * DI);
  float* dbc   = alloc((size_t)TT * 48);
  float* dtb   = alloc((size_t)TT * DI);
  float* yb    = alloc((size_t)TT * DI);
  float* xn    = alloc((size_t)TT * DM);
  float* cn    = alloc((size_t)TT * DM);
  float* qb    = alloc((size_t)TT * DM);
  float* kvb   = alloc((size_t)TT * 2 * DM);
  float* simb  = alloc((size_t)BB * LL * LL);
  float* pvb   = alloc((size_t)TT * DM);
  float* xa    = alloc((size_t)TT * DM);
  float* ff1   = alloc((size_t)TT * 2048);
  float* gg    = alloc((size_t)TT * 1024);
  float* zl    = alloc((size_t)TT);
  float* sout  = alloc((size_t)TT * 16);

  float* out  = (float*)d_out;
  float* klout = out + (size_t)TT * 22;

  // embeddings: pose_emb and shape_emb (-> hbuf, initial hidden)
  embed_kernel<<<dim3(TT, 4), 128, 0, stream>>>(P[0],
      P[2], P[3], P[5], P[6], P[8], P[9], P[11], P[12], pose_emb, hbuf);

  // ---------------- encoder ----------------
  StackW enc = { P[13], P[14], P[15], P[16], P[17], P[18], P[19], P[20], P[21], P[22], P[23] };
  run_stack(stream, pose_emb, hbuf, rbuf, hnbuf, xz, xc, dbc, dtb, yb, enc);

  const float* enc_ca[8] = { P[24], P[25], P[26], P[27], P[28], P[29], P[30], P[31] };
  run_attn(stream, hnbuf, pose_emb, xn, cn, qb, kvb, simb, pvb, xa, hnbuf, enc_ca); // + hidden
  const float* enc_ff[6] = { P[32], P[33], P[34], P[35], P[36], P[37] };
  run_ff(stream, xa, xn, ff1, gg, hbuf, enc_ff);                                    // + x

  // ---------------- latent ----------------
  zero_kernel<<<1, 32, 0, stream>>>(klout, BB);
  latent_kernel<<<(TT + 127) / 128, 128, 0, stream>>>(hbuf, P[38], P[39], P[40], P[41],
                                                      P[71], zl, klout);
  proj_kernel<<<(TT * DM + 255) / 256, 256, 0, stream>>>(zl, P[42], P[43], hbuf);

  // ---------------- decoder ----------------
  StackW dec = { P[44], P[45], P[46], P[47], P[48], P[49], P[50], P[51], P[52], P[53], P[54] };
  run_stack(stream, pose_emb, hbuf, rbuf, hnbuf, xz, xc, dbc, dtb, yb, dec);

  const float* dec_ca[8] = { P[55], P[56], P[57], P[58], P[59], P[60], P[61], P[62] };
  run_attn(stream, hnbuf, pose_emb, xn, cn, qb, kvb, simb, pvb, xa, nullptr, dec_ca); // lat
  const float* dec_ff[6] = { P[63], P[64], P[65], P[66], P[67], P[68] };
  run_ff(stream, xa, xn, ff1, gg, hbuf, dec_ff);                                      // lat + ff

  // head + assemble
  gemm(stream, hbuf, P[69], P[70], nullptr, sout, TT, 16, DM, DM, DM, 16, 0, 0, 0.f, 1, 0, 0, 0);
  assemble_kernel<<<(TT * 22 + 255) / 256, 256, 0, stream>>>(P[0], sout, out);
}
